// PtrNet1_3642132267467
// MI455X (gfx1250) — matrix operand order repaired and checked by reading, not compiled, not run on hardware
//
#include <hip/hip_runtime.h>
#include <hip/hip_bf16.h>

typedef __attribute__((ext_vector_type(16))) _Float16 v16h;
typedef __attribute__((ext_vector_type(8)))  _Float16 v8h;
typedef __attribute__((ext_vector_type(8)))  float    v8f;

#define B_SZ  512
#define N_SEQ 100
#define P_IN  20
#define E_DIM 256
#define H_DIM 512
#define G4H   2048
#define INF_  1e8f

__device__ __forceinline__ float sigmoidf_(float x) { return 1.f / (1.f + expf(-x)); }

// ---------------------------------------------------------------------------
// WMMA GEMM: C[M,N] = A1[M,K1] * Bp1 + (A2[M,K2] * Bp2) + bias
// A row-major f16 (lda in elements). B pre-packed into per-tile WMMA
// fragments: Bp[((kt*NT + nt)*32 + lane)*16 + j], K=kt*32+8*(lane>>4)+(j&7)+16*(j>>3),
// N=nt*16+(lane&15).
// One wave computes a 16x64 output tile: 4 independent accumulators reuse one
// A fragment per k-step (4 independent v_wmma chains pipeline with the loads).
// Requires M % 16 == 0, N % 64 == 0, K % 32 == 0.
// ---------------------------------------------------------------------------
__global__ void __launch_bounds__(256)
wmma_gemm_f16(const _Float16* __restrict__ A1, int lda1, int K1,
              const _Float16* __restrict__ Bp1,
              const _Float16* __restrict__ A2, int lda2, int K2,
              const _Float16* __restrict__ Bp2,
              const float* __restrict__ bias,
              float* __restrict__ Cf, _Float16* __restrict__ Ch, int ldc,
              int M, int N)
{
    const int lane = threadIdx.x & 31;
    const int wave = blockIdx.x * (blockDim.x >> 5) + (threadIdx.x >> 5);
    const int NT  = N >> 4;   // 16-wide fragment columns (packing granularity)
    const int NT4 = N >> 6;   // 64-wide wave tiles
    const int MT  = M >> 4;
    if (wave >= MT * NT4) return;         // wave-uniform: EXEC stays all-1s
    const int mt  = wave / NT4;
    const int nt4 = wave - mt * NT4;

    const int halfSel = lane >> 4;         // 0: lanes 0-15, 1: lanes 16-31
    const int mrow = (mt << 4) + (lane & 15);
    const int koff = halfSel << 3;         // A-fragment K sub-offset (0 or 8)
    const size_t bstride = (size_t)NT * 512; // fragment elems per k-tile row

    v8f acc[4] = {};

    {
        const _Float16* arow  = A1 + (size_t)mrow * lda1;
        const _Float16* bfrag = Bp1 + ((size_t)(nt4 << 2) * 32 + lane) * 16;
        for (int kt = 0; kt < (K1 >> 5); ++kt) {
            const v8h lo = *(const v8h*)(arow + (kt << 5) + koff);
            const v8h hi = *(const v8h*)(arow + (kt << 5) + koff + 16);
            v16h a;
#pragma unroll
            for (int i = 0; i < 8; ++i) { a[i] = lo[i]; a[8 + i] = hi[i]; }
            __builtin_prefetch((const void*)(bfrag + bstride), 0, 1);
#pragma unroll
            for (int s = 0; s < 4; ++s) {
                const v16h b = *(const v16h*)(bfrag + (size_t)s * 512);
                acc[s] = __builtin_amdgcn_wmma_f32_16x16x32_f16(
                    false, a, false, b, (short)0, acc[s], false, false);
            }
            bfrag += bstride;
        }
    }
    if (A2) {
        const _Float16* arow  = A2 + (size_t)mrow * lda2;
        const _Float16* bfrag = Bp2 + ((size_t)(nt4 << 2) * 32 + lane) * 16;
        for (int kt = 0; kt < (K2 >> 5); ++kt) {
            const v8h lo = *(const v8h*)(arow + (kt << 5) + koff);
            const v8h hi = *(const v8h*)(arow + (kt << 5) + koff + 16);
            v16h a;
#pragma unroll
            for (int i = 0; i < 8; ++i) { a[i] = lo[i]; a[8 + i] = hi[i]; }
            __builtin_prefetch((const void*)(bfrag + bstride), 0, 1);
#pragma unroll
            for (int s = 0; s < 4; ++s) {
                const v16h b = *(const v16h*)(bfrag + (size_t)s * 512);
                acc[s] = __builtin_amdgcn_wmma_f32_16x16x32_f16(
                    false, a, false, b, (short)0, acc[s], false, false);
            }
            bfrag += bstride;
        }
    }

    // C/D layout: VGPR r: lanes 0-15 -> M=r, lanes 16-31 -> M=8+r; N=lane&15
    const int rbase = (mt << 4) + (halfSel << 3);
#pragma unroll
    for (int s = 0; s < 4; ++s) {
        const int cn = (nt4 << 6) + (s << 4) + (lane & 15);
        const float bv = bias ? bias[cn] : 0.f;
        if (Cf) {
#pragma unroll
            for (int r = 0; r < 8; ++r)
                Cf[(size_t)(rbase + r) * ldc + cn] = acc[s][r] + bv;
        }
        if (Ch) {
#pragma unroll
            for (int r = 0; r < 8; ++r)
                Ch[(size_t)(rbase + r) * ldc + cn] = (_Float16)(acc[s][r] + bv);
        }
    }
}

// Repack weight W[N,Kreal] (row-major, Linear convention) into WMMA B
// fragments with K padded to Kpad (multiple of 32).
__global__ void pack_weight_f16(const float* __restrict__ W,
                                _Float16* __restrict__ Bp,
                                int N, int Kpad, int Kreal)
{
    int idx = blockIdx.x * blockDim.x + threadIdx.x;
    int total = N * Kpad;
    if (idx >= total) return;
    int j    = idx & 15;
    int lane = (idx >> 4) & 31;
    int tile = idx >> 9;
    int NT = N >> 4;
    int kt = tile / NT;
    int nt = tile - kt * NT;
    int n = (nt << 4) + (lane & 15);
    int k = (kt << 5) + ((lane >> 4) << 3) + (j & 7) + ((j >> 3) << 4);
    float v = (k < Kreal) ? W[(size_t)n * Kreal + k] : 0.f;
    Bp[idx] = (_Float16)v;
}

__global__ void pad_x_f16(const float* __restrict__ x, _Float16* __restrict__ xp,
                          int rows)
{
    int idx = blockIdx.x * blockDim.x + threadIdx.x;
    if (idx >= rows * 32) return;
    int r = idx >> 5, p = idx & 31;
    xp[idx] = (p < P_IN) ? (_Float16)x[(size_t)r * P_IN + p] : (_Float16)0.f;
}

__global__ void bias_sum_k(const float* __restrict__ a, const float* __restrict__ b,
                           float* __restrict__ o, int n)
{
    int i = blockIdx.x * blockDim.x + threadIdx.x;
    if (i < n) o[i] = a[i] + b[i];
}

__global__ void fill_f32_k(float* p, float v, int n)
{
    int i = blockIdx.x * blockDim.x + threadIdx.x;
    if (i < n) p[i] = v;
}

__global__ void fill_f16_k(_Float16* p, float v, int n)
{
    int i = blockIdx.x * blockDim.x + threadIdx.x;
    if (i < n) p[i] = (_Float16)v;
}

__global__ void init_decin_k(const float* __restrict__ d0, _Float16* __restrict__ decin)
{
    int i = blockIdx.x * blockDim.x + threadIdx.x;
    if (i < B_SZ * E_DIM) decin[i] = (_Float16)d0[i & (E_DIM - 1)];
}

__global__ void write_pi_k(const int* __restrict__ y, float* __restrict__ o, int n)
{
    int i = blockIdx.x * blockDim.x + threadIdx.x;
    if (i < n) o[i] = (float)y[i];
}

// LSTM gates (PyTorch order i,f,g,o). Updates c, writes h as f16 (contiguous)
// and optionally a strided f16 copy (encoder outputs -> ref buffer).
__global__ void lstm_gates_k(const float* __restrict__ g, float* __restrict__ c,
                             _Float16* __restrict__ h16, _Float16* __restrict__ href,
                             int strideRef)
{
    int idx = blockIdx.x * blockDim.x + threadIdx.x;
    if (idx >= B_SZ * H_DIM) return;
    int b  = idx >> 9;
    int hh = idx & (H_DIM - 1);
    const float* gr = g + (size_t)b * G4H;
    float iv = sigmoidf_(gr[hh]);
    float fv = sigmoidf_(gr[H_DIM + hh]);
    float gv = tanhf(gr[2 * H_DIM + hh]);
    float ov = sigmoidf_(gr[3 * H_DIM + hh]);
    float cc = fv * c[idx] + iv * gv;
    c[idx] = cc;
    float hv = ov * tanhf(cc);
    h16[idx] = (_Float16)hv;
    if (href) href[(size_t)b * strideRef + hh] = (_Float16)hv;
}

// u[b,n] = Vec . tanh(q[b,:] + u2[b,n,:]); one wave per (b,n), shuffle-reduce.
__global__ void attn_scores_k(const float* __restrict__ q, const float* __restrict__ u2,
                              const float* __restrict__ vec, float* __restrict__ u,
                              int BN, int H_, int N_)
{
    int wave = blockIdx.x * (blockDim.x >> 5) + (threadIdx.x >> 5);
    int lane = threadIdx.x & 31;
    if (wave >= BN) return;
    int b = wave / N_;
    const float* qr = q + (size_t)b * H_;
    const float* ur = u2 + (size_t)wave * H_;
    float s = 0.f;
    for (int hh = lane; hh < H_; hh += 32)
        s += vec[hh] * tanhf(qr[hh] + ur[hh]);
#pragma unroll
    for (int o = 16; o; o >>= 1) s += __shfl_xor(s, o, 32);
    if (lane == 0) u[wave] = s;
}

// a[b,:] = softmax(u[b,:] - INF*mask[b,:]); one wave per batch row.
__global__ void softmax_rows_k(const float* __restrict__ u, const float* __restrict__ mask,
                               float* __restrict__ a, int B_, int N_)
{
    int wave = blockIdx.x * (blockDim.x >> 5) + (threadIdx.x >> 5);
    int lane = threadIdx.x & 31;
    if (wave >= B_) return;
    const float* ur = u + (size_t)wave * N_;
    const float* mr = mask + (size_t)wave * N_;
    float mx = -1e30f;
    for (int n = lane; n < N_; n += 32)
        mx = fmaxf(mx, ur[n] - INF_ * mr[n]);
#pragma unroll
    for (int o = 16; o; o >>= 1) mx = fmaxf(mx, __shfl_xor(mx, o, 32));
    float s = 0.f;
    for (int n = lane; n < N_; n += 32)
        s += expf(ur[n] - INF_ * mr[n] - mx);
#pragma unroll
    for (int o = 16; o; o >>= 1) s += __shfl_xor(s, o, 32);
    float inv = 1.f / s;
    for (int n = lane; n < N_; n += 32)
        a[(size_t)wave * N_ + n] = expf(ur[n] - INF_ * mr[n] - mx) * inv;
}

// q16[b,h] = sum_n a[b,n] * ref[b,n,h]
__global__ void weighted_sum_k(const float* __restrict__ a, const _Float16* __restrict__ ref,
                               _Float16* __restrict__ q16, int B_, int N_, int H_)
{
    int idx = blockIdx.x * blockDim.x + threadIdx.x;
    if (idx >= B_ * H_) return;
    int b  = idx / H_;
    int hh = idx - b * H_;
    float s = 0.f;
    const float* ar = a + (size_t)b * N_;
    const _Float16* rr = ref + ((size_t)b * N_) * H_ + hh;
    for (int n = 0; n < N_; ++n)
        s += ar[n] * (float)rr[(size_t)n * H_];
    q16[idx] = (_Float16)s;
}

// Final pointer logits for step t: clip, log_softmax, write ps row, accumulate
// ll at y[b,t], update mask, gather next decoder input. One wave per batch.
__global__ void dec_finish_k(const float* __restrict__ u2, const int* __restrict__ y,
                             const _Float16* __restrict__ embed16,
                             float* __restrict__ mask, float* __restrict__ ll,
                             float* __restrict__ ps, _Float16* __restrict__ decin, int t)
{
    int wave = blockIdx.x * (blockDim.x >> 5) + (threadIdx.x >> 5);
    int lane = threadIdx.x & 31;
    if (wave >= B_SZ) return;
    int b = wave;
    const float* ur = u2 + (size_t)b * N_SEQ;
    const float* mr = mask + (size_t)b * N_SEQ;
    float mx = -1e30f;
    for (int n = lane; n < N_SEQ; n += 32)
        mx = fmaxf(mx, 10.f * tanhf(ur[n]) - INF_ * mr[n]);
#pragma unroll
    for (int o = 16; o; o >>= 1) mx = fmaxf(mx, __shfl_xor(mx, o, 32));
    float s = 0.f;
    for (int n = lane; n < N_SEQ; n += 32)
        s += expf(10.f * tanhf(ur[n]) - INF_ * mr[n] - mx);
#pragma unroll
    for (int o = 16; o; o >>= 1) s += __shfl_xor(s, o, 32);
    float lse = mx + logf(s);
    float* pr = ps + ((size_t)b * N_SEQ + t) * N_SEQ;
    for (int n = lane; n < N_SEQ; n += 32)
        pr[n] = 10.f * tanhf(ur[n]) - INF_ * mr[n] - lse;
    int yi = y[(size_t)b * N_SEQ + t];
    if (lane == 0) {
        ll[b] += 10.f * tanhf(ur[yi]) - INF_ * mr[yi] - lse;
        mask[(size_t)b * N_SEQ + yi] += 1.f;
    }
    const _Float16* er = embed16 + ((size_t)b * N_SEQ + yi) * E_DIM;
    _Float16* dr = decin + (size_t)b * E_DIM;
    for (int e = lane; e < E_DIM; e += 32) dr[e] = er[e];
}

// ---------------------------------------------------------------------------
extern "C" void kernel_launch(void* const* d_in, const int* in_sizes, int n_in,
                              void* d_out, int out_size, void* d_ws, size_t ws_size,
                              hipStream_t stream)
{
    (void)in_sizes; (void)n_in; (void)out_size; (void)ws_size;

    const float* x        = (const float*)d_in[0];
    const int*   y        = (const int*)d_in[1];
    const float* emb_W    = (const float*)d_in[2];
    const float* enc_Wih  = (const float*)d_in[3];
    const float* enc_Whh  = (const float*)d_in[4];
    const float* enc_bih  = (const float*)d_in[5];
    const float* enc_bhh  = (const float*)d_in[6];
    const float* dec_Wih  = (const float*)d_in[7];
    const float* dec_Whh  = (const float*)d_in[8];
    const float* dec_bih  = (const float*)d_in[9];
    const float* dec_bhh  = (const float*)d_in[10];
    const float* Vec      = (const float*)d_in[11];
    const float* Vec2     = (const float*)d_in[12];
    const float* Wq_W     = (const float*)d_in[13];
    const float* Wq_b     = (const float*)d_in[14];
    const float* Wref_W   = (const float*)d_in[15];
    const float* Wref_b   = (const float*)d_in[16];
    const float* Wq2_W    = (const float*)d_in[17];
    const float* Wq2_b    = (const float*)d_in[18];
    const float* Wref2_W  = (const float*)d_in[19];
    const float* Wref2_b  = (const float*)d_in[20];
    const float* dec_in0  = (const float*)d_in[21];

    float* out    = (float*)d_out;
    float* out_pi = out;                         // [B*N]
    float* out_ll = out + (size_t)B_SZ * N_SEQ;  // [B]
    float* out_ps = out_ll + B_SZ;               // [B,N,N]

    char* ws = (char*)d_ws;
    size_t off = 0;
    auto alloc = [&](size_t bytes) -> void* {
        off = (off + 255) & ~(size_t)255;
        void* p = (void*)(ws + off);
        off += bytes;
        return p;
    };
    const size_t BN = (size_t)B_SZ * N_SEQ;

    _Float16* xpad     = (_Float16*)alloc(BN * 32 * 2);
    _Float16* embed16  = (_Float16*)alloc(BN * E_DIM * 2);
    _Float16* packEmb  = (_Float16*)alloc((size_t)E_DIM * 32 * 2);
    _Float16* packEWih = (_Float16*)alloc((size_t)G4H * E_DIM * 2);
    _Float16* packEWhh = (_Float16*)alloc((size_t)G4H * H_DIM * 2);
    _Float16* packDWih = (_Float16*)alloc((size_t)G4H * E_DIM * 2);
    _Float16* packDWhh = (_Float16*)alloc((size_t)G4H * H_DIM * 2);
    _Float16* packWq   = (_Float16*)alloc((size_t)H_DIM * H_DIM * 2);
    _Float16* packWref = (_Float16*)alloc((size_t)H_DIM * H_DIM * 2);
    _Float16* packWq2  = (_Float16*)alloc((size_t)H_DIM * H_DIM * 2);
    _Float16* packWrf2 = (_Float16*)alloc((size_t)H_DIM * H_DIM * 2);
    float*    encBsum  = (float*)alloc((size_t)G4H * 4);
    float*    decBsum  = (float*)alloc((size_t)G4H * 4);
    float*    gbuf     = (float*)alloc((size_t)B_SZ * G4H * 4);
    _Float16* h16      = (_Float16*)alloc((size_t)B_SZ * H_DIM * 2);
    float*    cbuf     = (float*)alloc((size_t)B_SZ * H_DIM * 4);
    _Float16* ref16    = (_Float16*)alloc(BN * H_DIM * 2);
    float*    u2g      = (float*)alloc(BN * H_DIM * 4);
    float*    u2p      = (float*)alloc(BN * H_DIM * 4);
    float*    q1       = (float*)alloc((size_t)B_SZ * H_DIM * 4);
    float*    q2       = (float*)alloc((size_t)B_SZ * H_DIM * 4);
    _Float16* q16      = (_Float16*)alloc((size_t)B_SZ * H_DIM * 2);
    float*    uscore   = (float*)alloc(BN * 4);
    float*    aattn    = (float*)alloc(BN * 4);
    float*    maskb    = (float*)alloc(BN * 4);
    _Float16* decin16  = (_Float16*)alloc((size_t)B_SZ * E_DIM * 2);

    auto ew = [](int n) { return dim3((n + 255) / 256); };
    auto wv = [](int waves) { return dim3((waves + 7) / 8); };   // 8 waves/block
    auto gemmg = [](int M, int N) { return dim3(((M >> 4) * (N >> 6) + 7) / 8); };
    const dim3 TB(256);

    // ---- one-time packs / conversions -------------------------------------
    pack_weight_f16<<<ew(E_DIM * 32), TB, 0, stream>>>(emb_W, packEmb, E_DIM, 32, P_IN);
    pack_weight_f16<<<ew(G4H * E_DIM), TB, 0, stream>>>(enc_Wih, packEWih, G4H, E_DIM, E_DIM);
    pack_weight_f16<<<ew(G4H * H_DIM), TB, 0, stream>>>(enc_Whh, packEWhh, G4H, H_DIM, H_DIM);
    pack_weight_f16<<<ew(G4H * E_DIM), TB, 0, stream>>>(dec_Wih, packDWih, G4H, E_DIM, E_DIM);
    pack_weight_f16<<<ew(G4H * H_DIM), TB, 0, stream>>>(dec_Whh, packDWhh, G4H, H_DIM, H_DIM);
    pack_weight_f16<<<ew(H_DIM * H_DIM), TB, 0, stream>>>(Wq_W, packWq, H_DIM, H_DIM, H_DIM);
    pack_weight_f16<<<ew(H_DIM * H_DIM), TB, 0, stream>>>(Wref_W, packWref, H_DIM, H_DIM, H_DIM);
    pack_weight_f16<<<ew(H_DIM * H_DIM), TB, 0, stream>>>(Wq2_W, packWq2, H_DIM, H_DIM, H_DIM);
    pack_weight_f16<<<ew(H_DIM * H_DIM), TB, 0, stream>>>(Wref2_W, packWrf2, H_DIM, H_DIM, H_DIM);
    bias_sum_k<<<ew(G4H), TB, 0, stream>>>(enc_bih, enc_bhh, encBsum, G4H);
    bias_sum_k<<<ew(G4H), TB, 0, stream>>>(dec_bih, dec_bhh, decBsum, G4H);
    pad_x_f16<<<ew((int)BN * 32), TB, 0, stream>>>(x, xpad, (int)BN);

    // embed = x @ emb_W.T  -> f16  [B*N, E]
    wmma_gemm_f16<<<gemmg((int)BN, E_DIM), TB, 0, stream>>>(
        xpad, 32, 32, packEmb, nullptr, 0, 0, nullptr,
        nullptr, nullptr, embed16, E_DIM, (int)BN, E_DIM);

    // ---- state init --------------------------------------------------------
    fill_f16_k<<<ew(B_SZ * H_DIM), TB, 0, stream>>>(h16, 0.f, B_SZ * H_DIM);
    fill_f32_k<<<ew(B_SZ * H_DIM), TB, 0, stream>>>(cbuf, 0.f, B_SZ * H_DIM);
    fill_f32_k<<<ew((int)BN), TB, 0, stream>>>(maskb, 0.f, (int)BN);
    fill_f32_k<<<ew(B_SZ), TB, 0, stream>>>(out_ll, 0.f, B_SZ);
    init_decin_k<<<ew(B_SZ * E_DIM), TB, 0, stream>>>(dec_in0, decin16);
    write_pi_k<<<ew((int)BN), TB, 0, stream>>>(y, out_pi, (int)BN);

    // ---- encoder LSTM ------------------------------------------------------
    for (int t = 0; t < N_SEQ; ++t) {
        wmma_gemm_f16<<<gemmg(B_SZ, G4H), TB, 0, stream>>>(
            embed16 + (size_t)t * E_DIM, N_SEQ * E_DIM, E_DIM, packEWih,
            h16, H_DIM, H_DIM, packEWhh,
            encBsum, gbuf, nullptr, G4H, B_SZ, G4H);
        lstm_gates_k<<<ew(B_SZ * H_DIM), TB, 0, stream>>>(
            gbuf, cbuf, h16, ref16 + (size_t)t * H_DIM, N_SEQ * H_DIM);
    }

    // ---- u2g / u2p projections of ref -------------------------------------
    wmma_gemm_f16<<<gemmg((int)BN, H_DIM), TB, 0, stream>>>(
        ref16, H_DIM, H_DIM, packWref, nullptr, 0, 0, nullptr,
        Wref_b, u2g, nullptr, H_DIM, (int)BN, H_DIM);
    wmma_gemm_f16<<<gemmg((int)BN, H_DIM), TB, 0, stream>>>(
        ref16, H_DIM, H_DIM, packWrf2, nullptr, 0, 0, nullptr,
        Wref2_b, u2p, nullptr, H_DIM, (int)BN, H_DIM);

    // ---- decoder with attention -------------------------------------------
    for (int t = 0; t < N_SEQ; ++t) {
        wmma_gemm_f16<<<gemmg(B_SZ, G4H), TB, 0, stream>>>(
            decin16, E_DIM, E_DIM, packDWih,
            h16, H_DIM, H_DIM, packDWhh,
            decBsum, gbuf, nullptr, G4H, B_SZ, G4H);
        lstm_gates_k<<<ew(B_SZ * H_DIM), TB, 0, stream>>>(
            gbuf, cbuf, h16, nullptr, 0);

        // glimpse
        wmma_gemm_f16<<<gemmg(B_SZ, H_DIM), TB, 0, stream>>>(
            h16, H_DIM, H_DIM, packWq, nullptr, 0, 0, nullptr,
            Wq_b, q1, nullptr, H_DIM, B_SZ, H_DIM);
        attn_scores_k<<<wv((int)BN), TB, 0, stream>>>(q1, u2g, Vec, uscore,
                                                      (int)BN, H_DIM, N_SEQ);
        softmax_rows_k<<<wv(B_SZ), TB, 0, stream>>>(uscore, maskb, aattn, B_SZ, N_SEQ);
        weighted_sum_k<<<ew(B_SZ * H_DIM), TB, 0, stream>>>(aattn, ref16, q16,
                                                            B_SZ, N_SEQ, H_DIM);

        // pointer logits
        wmma_gemm_f16<<<gemmg(B_SZ, H_DIM), TB, 0, stream>>>(
            q16, H_DIM, H_DIM, packWq2, nullptr, 0, 0, nullptr,
            Wq2_b, q2, nullptr, H_DIM, B_SZ, H_DIM);
        attn_scores_k<<<wv((int)BN), TB, 0, stream>>>(q2, u2p, Vec2, uscore,
                                                      (int)BN, H_DIM, N_SEQ);
        dec_finish_k<<<wv(B_SZ), TB, 0, stream>>>(uscore, y, embed16, maskb,
                                                  out_ll, out_ps, decin16, t);
    }
}
